// DSW_7773890806423
// MI455X (gfx1250) — matrix operand order repaired and checked by reading, not compiled, not run on hardware
//
#include <hip/hip_runtime.h>

typedef __attribute__((ext_vector_type(16))) _Float16 v16h;
typedef __attribute__((ext_vector_type(8)))  _Float16 v8h;
typedef __attribute__((ext_vector_type(2)))  _Float16 h2;
typedef __attribute__((ext_vector_type(8)))  float    v8f;

__device__ __forceinline__ int iclamp(int v, int lo, int hi) {
  return v < lo ? lo : (v > hi ? hi : v);
}

__device__ __forceinline__ v16h frag_from_lds_A(const _Float16* row, int hi) {
  // A 16x32 f16 layout: half0 e0..7->K0..7, e8..15->K16..23; half1 +8
  v8h a0 = *(const v8h*)(row + 8 * hi);
  v8h a1 = *(const v8h*)(row + 16 + 8 * hi);
  v16h a;
#pragma unroll
  for (int e = 0; e < 8; ++e) { a[e] = a0[e]; a[e + 8] = a1[e]; }
  return a;
}
__device__ __forceinline__ v16h frag_from_lds_B(const _Float16* row, int hi) {
  // B 32x16 f16 layout: half0 e0..7->K0..7, e8..15->K16..23; half1 +8
  v8h b0 = *(const v8h*)(row + 8 * hi);
  v8h b1 = *(const v8h*)(row + 16 + 8 * hi);
  v16h b;
#pragma unroll
  for (int e = 0; e < 8; ++e) { b[e] = b0[e]; b[e + 8] = b1[e]; }
  return b;
}

// ---------------------------------------------------------------------------
// Generic 3x3 conv (pad=1), implicit GEMM on WMMA f16->f32.
// WG = 256 thr = 8 waves as 4(M) x 2(N); WG tile = 64 px x 64 ch.
// Double-buffered LDS, software-pipelined global loads, float-mask predication
// (no pipelined bool lane-masks), packed 2xf16 LDS stores.
__global__ __launch_bounds__(256) void conv3x3_wmma(
    const float* __restrict__ in, const float* __restrict__ wgt,
    const float* __restrict__ bias, const float* __restrict__ resid,
    float* __restrict__ out, int Cin, int Cout, int H, int W, int relu) {
  __shared__ _Float16 As[2][64][40];
  __shared__ _Float16 Bs[2][64][40];
  __shared__ float    Cs[64][72];

  const int HW = H * W;
  const int K = Cin * 9;
  const int img = blockIdx.z;
  const int t = threadIdx.x;
  const int lane = t & 31, wid = t >> 5;
  const int hi = lane >> 4, l15 = lane & 15;
  const int mw = wid >> 1, nw = wid & 1;   // 4 x 2 wave grid
  const int m0 = blockIdx.x * 64;
  const int n0 = blockIdx.y * 64;

  const float* xin = in + (size_t)img * Cin * HW;

  // A staging: pixel ms = t&63 (coalesced); k pairs 8i + 2*kp + j
  const int ms = t & 63;
  const int kp = t >> 6;                   // 0..3
  const int gm = m0 + ms;
  const bool gmv = gm < HW;
  const int gy = gmv ? gm / W : 0;
  const int gx = gmv ? gm % W : 0;
  // B staging: ch rows nb4 + 16i; k pairs kp2 + j
  const int kp2 = (t & 15) * 2;
  const int nb4 = t >> 4;                  // 0..15

  float areg[4][2], amask[4][2];
  float breg[4][2], bmask[4][2];

  auto issueA = [&](int kk) {
#pragma unroll
    for (int i = 0; i < 4; ++i)
#pragma unroll
      for (int j = 0; j < 2; ++j) {
        int ka = kk + 8 * i + 2 * kp + j;
        int kaa = ka < K ? ka : 0;
        int c = kaa / 9;
        int tap = kaa - 9 * c;
        int yy = gy + tap / 3 - 1;
        int xx = gx + (tap % 3) - 1;
        bool ok = gmv & (ka < K) & (yy >= 0) & (yy < H) & (xx >= 0) & (xx < W);
        amask[i][j] = ok ? 1.f : 0.f;
        areg[i][j] = xin[(size_t)c * HW + iclamp(yy, 0, H - 1) * W +
                         iclamp(xx, 0, W - 1)];   // always in-bounds
      }
  };
  auto issueB = [&](int kk) {
#pragma unroll
    for (int i = 0; i < 4; ++i)
#pragma unroll
      for (int j = 0; j < 2; ++j) {
        int nl = nb4 + 16 * i;
        int kb = kk + kp2 + j;
        bool ok = ((n0 + nl) < Cout) & (kb < K);
        bmask[i][j] = ok ? 1.f : 0.f;
        int nc = (n0 + nl) < Cout ? (n0 + nl) : 0;
        int kc = kb < K ? kb : 0;
        breg[i][j] = wgt[(size_t)nc * K + kc];
      }
  };

  v8f acc0 = {0.f, 0.f, 0.f, 0.f, 0.f, 0.f, 0.f, 0.f};
  v8f acc1 = acc0;

  issueA(0);
  issueB(0);
  int cur = 0;
  for (int kk = 0; kk < K; kk += 32) {
    // consume staged registers -> LDS buf[cur] (packed 2xf16 stores)
#pragma unroll
    for (int i = 0; i < 4; ++i) {
      h2 pa = {(_Float16)(areg[i][0] * amask[i][0]),
               (_Float16)(areg[i][1] * amask[i][1])};
      *(h2*)&As[cur][ms][8 * i + 2 * kp] = pa;
      h2 pb = {(_Float16)(breg[i][0] * bmask[i][0]),
               (_Float16)(breg[i][1] * bmask[i][1])};
      *(h2*)&Bs[cur][nb4 + 16 * i][kp2] = pb;
    }
    // prefetch next k-step while this one computes
    if (kk + 32 < K) { issueA(kk + 32); issueB(kk + 32); }
    __syncthreads();
    v16h a = frag_from_lds_A(&As[cur][mw * 16 + l15][0], hi);
    v16h b0f = frag_from_lds_B(&Bs[cur][nw * 32 + l15][0], hi);
    v16h b1f = frag_from_lds_B(&Bs[cur][nw * 32 + 16 + l15][0], hi);
    acc0 = __builtin_amdgcn_wmma_f32_16x16x32_f16(false, a, false, b0f,
                                                  (short)0, acc0, false, false);
    acc1 = __builtin_amdgcn_wmma_f32_16x16x32_f16(false, a, false, b1f,
                                                  (short)0, acc1, false, false);
    cur ^= 1;
  }

  // epilogue: acc -> LDS -> coalesced fused store
  __syncthreads();
#pragma unroll
  for (int r = 0; r < 8; ++r) {
    Cs[mw * 16 + r + 8 * hi][nw * 32 + l15] = acc0[r];
    Cs[mw * 16 + r + 8 * hi][nw * 32 + 16 + l15] = acc1[r];
  }
  __syncthreads();

#pragma unroll
  for (int i = 0; i < 16; ++i) {
    int nl = kp + 4 * i;
    int n = n0 + nl;
    int m = m0 + ms;
    if (n < Cout && m < HW) {
      float v = Cs[ms][nl];
      if (bias) v += bias[n];
      if (resid) v += resid[(size_t)img * Cout * HW + (size_t)n * HW + m];
      if (relu) v = fmaxf(v, 0.f);
      out[(size_t)img * Cout * HW + (size_t)n * HW + m] = v;
    }
  }
}

// ---------------------------------------------------------------------------
// Deformable 3x3 conv (dg=8 groups, 9 taps), K = 576.
// WG tile 32 px x 64 ch (8 waves as 2x4, 1 WMMA each); double-buffered LDS;
// offset + weight loads pipelined; bilinear corner loads batched per step.
__global__ __launch_bounds__(256) void dconv3x3_wmma(
    const float* __restrict__ in, const float* __restrict__ off,
    const float* __restrict__ wgt, const float* __restrict__ bias,
    float* __restrict__ out, int H, int W) {
  __shared__ _Float16 As[2][32][40];
  __shared__ _Float16 Bs[2][64][40];
  __shared__ float    Cs[32][72];

  const int Cout = 64;
  const int HW = H * W;
  const int K = 576;
  const int img = blockIdx.z;
  const int t = threadIdx.x;
  const int lane = t & 31, wid = t >> 5;
  const int hi = lane >> 4, l15 = lane & 15;
  const int mw = wid >> 2, nw = wid & 3;
  const int m0 = blockIdx.x * 32;
  const int n0 = blockIdx.y * 64;

  const float* xin = in + (size_t)img * 64 * HW;
  const float* offp = off + (size_t)img * 144 * HW;

  // A staging: pixel ms = t&31; 4 elems at k = 2*kq + 16*i + j (i<2, j<2)
  const int ms = t & 31;
  const int kq = t >> 5;                   // 0..7
  const int gm = m0 + ms;
  const bool gmv = gm < HW;
  const int gy = gmv ? gm / W : 0;
  const int gx = gmv ? gm % W : 0;
  const float gmask = gmv ? 1.f : 0.f;
  // B staging: ch rows nb4 + 16i; k pairs kp2 + j
  const int kp2 = (t & 15) * 2;
  const int nb4 = t >> 4;                  // 0..15

  float oyr[2][2], oxr[2][2];
  float breg[4][2];

  auto issueOff = [&](int kk) {
#pragma unroll
    for (int i = 0; i < 2; ++i)
#pragma unroll
      for (int j = 0; j < 2; ++j) {
        int ka = kk + 2 * kq + 16 * i + j;
        int c = ka / 9;
        int tap = ka - 9 * c;
        int g = c >> 3;
        oyr[i][j] = offp[(size_t)(g * 18 + tap * 2 + 0) * HW + gm];
        oxr[i][j] = offp[(size_t)(g * 18 + tap * 2 + 1) * HW + gm];
      }
  };
  auto issueB = [&](int kk) {
#pragma unroll
    for (int i = 0; i < 4; ++i)
#pragma unroll
      for (int j = 0; j < 2; ++j) {
        int nl = nb4 + 16 * i;
        int kb = kk + kp2 + j;
        int nc = (n0 + nl) < Cout ? (n0 + nl) : 0;
        breg[i][j] = wgt[(size_t)nc * K + kb];
      }
  };

  v8f acc = {0.f, 0.f, 0.f, 0.f, 0.f, 0.f, 0.f, 0.f};

  issueOff(0);
  issueB(0);
  int cur = 0;
  for (int kk = 0; kk < K; kk += 32) {
    // ---- A: bilinear sampling; compute all 16 corner addrs, batch loads ----
    const float* ap[16];
    float avm[16];
    float wyv[2][2], wxv[2][2];
#pragma unroll
    for (int i = 0; i < 2; ++i)
#pragma unroll
      for (int j = 0; j < 2; ++j) {
        int e = 2 * i + j;
        int ka = kk + 2 * kq + 16 * i + j;
        int c = ka / 9;
        int tap = ka - 9 * c;
        float sy = oyr[i][j] + (float)(tap / 3 - 1) + (float)gy;
        float sx = oxr[i][j] + (float)((tap % 3) - 1) + (float)gx;
        float y0f = floorf(sy), x0f = floorf(sx);
        wyv[i][j] = sy - y0f;
        wxv[i][j] = sx - x0f;
        int y0 = (int)y0f, x0 = (int)x0f;
        int y0c = iclamp(y0, 0, H - 1), y1c = iclamp(y0 + 1, 0, H - 1);
        int x0c = iclamp(x0, 0, W - 1), x1c = iclamp(x0 + 1, 0, W - 1);
        const float* ch = xin + (size_t)c * HW;
        ap[4 * e + 0] = ch + y0c * W + x0c;
        ap[4 * e + 1] = ch + y0c * W + x1c;
        ap[4 * e + 2] = ch + y1c * W + x0c;
        ap[4 * e + 3] = ch + y1c * W + x1c;
        float vy0 = ((y0 >= 0) & (y0 < H)) ? 1.f : 0.f;
        float vy1 = ((y0 + 1 >= 0) & (y0 + 1 < H)) ? 1.f : 0.f;
        float vx0 = ((x0 >= 0) & (x0 < W)) ? 1.f : 0.f;
        float vx1 = ((x0 + 1 >= 0) & (x0 + 1 < W)) ? 1.f : 0.f;
        avm[4 * e + 0] = vy0 * vx0;
        avm[4 * e + 1] = vy0 * vx1;
        avm[4 * e + 2] = vy1 * vx0;
        avm[4 * e + 3] = vy1 * vx1;
      }
    float cr[16];
#pragma unroll
    for (int q = 0; q < 16; ++q) cr[q] = *ap[q];   // batched load clause
#pragma unroll
    for (int i = 0; i < 2; ++i) {
      _Float16 pk[2];
#pragma unroll
      for (int j = 0; j < 2; ++j) {
        int e = 2 * i + j;
        float wy = wyv[i][j], wx = wxv[i][j];
        float val = cr[4 * e + 0] * avm[4 * e + 0] * (1.f - wy) * (1.f - wx) +
                    cr[4 * e + 1] * avm[4 * e + 1] * (1.f - wy) * wx +
                    cr[4 * e + 2] * avm[4 * e + 2] * wy * (1.f - wx) +
                    cr[4 * e + 3] * avm[4 * e + 3] * wy * wx;
        pk[j] = (_Float16)(val * gmask);
      }
      h2 pa = {pk[0], pk[1]};
      *(h2*)&As[cur][ms][2 * kq + 16 * i] = pa;
    }
#pragma unroll
    for (int i = 0; i < 4; ++i) {
      h2 pb = {(_Float16)breg[i][0], (_Float16)breg[i][1]};
      *(h2*)&Bs[cur][nb4 + 16 * i][kp2] = pb;
    }
    if (kk + 32 < K) { issueOff(kk + 32); issueB(kk + 32); }
    __syncthreads();
    v16h a = frag_from_lds_A(&As[cur][mw * 16 + l15][0], hi);
    v16h b = frag_from_lds_B(&Bs[cur][nw * 16 + l15][0], hi);
    acc = __builtin_amdgcn_wmma_f32_16x16x32_f16(false, a, false, b, (short)0,
                                                 acc, false, false);
    cur ^= 1;
  }

  __syncthreads();
#pragma unroll
  for (int r = 0; r < 8; ++r)
    Cs[mw * 16 + r + 8 * hi][nw * 16 + l15] = acc[r];
  __syncthreads();

#pragma unroll
  for (int i = 0; i < 8; ++i) {
    int nl = kq + 8 * i;
    int n = n0 + nl;
    int m = m0 + ms;
    if (n < Cout && m < HW)
      out[(size_t)img * Cout * HW + (size_t)n * HW + m] = Cs[ms][nl] + bias[n];
  }
}

// ---------------------------------------------------------------------------
// Gather / layout kernels
__global__ void build_cat_kernel(const float* __restrict__ feats,
                                 float* __restrict__ cat, int HW) {
  size_t total = (size_t)16 * 128 * HW;
  size_t idx = (size_t)blockIdx.x * blockDim.x + threadIdx.x;
  if (idx >= total) return;
  int i = (int)(idx / ((size_t)128 * HW));
  size_t rem = idx - (size_t)i * 128 * HW;
  int c = (int)(rem / HW);
  int m = (int)(rem - (size_t)c * HW);
  int b = i >> 2, slot = i & 3;
  int f = slot + (slot >= 2 ? 1 : 0);
  float v;
  if (c < 64)
    v = feats[((size_t)(b * 5 + 2) * 64 + c) * HW + m];
  else
    v = feats[((size_t)(b * 5 + f) * 64 + (c - 64)) * HW + m];
  cat[idx] = v;
}

__global__ void build_supp_kernel(const float* __restrict__ feats,
                                  float* __restrict__ supp, int HW) {
  size_t total = (size_t)16 * 64 * HW;
  size_t idx = (size_t)blockIdx.x * blockDim.x + threadIdx.x;
  if (idx >= total) return;
  int i = (int)(idx / ((size_t)64 * HW));
  size_t rem = idx - (size_t)i * 64 * HW;
  int b = i >> 2, slot = i & 3;
  int f = slot + (slot >= 2 ? 1 : 0);
  supp[idx] = feats[((size_t)(b * 5 + f) * 64) * HW + rem];
}

__global__ void build_lrs_kernel(const float* __restrict__ im,
                                 const float* __restrict__ x,
                                 float* __restrict__ lrs, int HW) {
  size_t total = (size_t)4 * 5 * 3 * HW;
  size_t idx = (size_t)blockIdx.x * blockDim.x + threadIdx.x;
  if (idx >= total) return;
  int b = (int)(idx / ((size_t)15 * HW));
  size_t rem = idx - (size_t)b * 15 * HW;
  int f = (int)(rem / ((size_t)3 * HW));
  size_t rem2 = rem - (size_t)f * 3 * HW;  // c*HW + m
  float v;
  if (f == 2) {
    v = x[((size_t)(b * 5 + 2) * 3) * HW + rem2];
  } else {
    int j = (f < 2) ? f : (f - 1);
    v = im[((size_t)(b * 4 + j) * 3) * HW + rem2];
  }
  lrs[idx] = v;
}

// pixel shuffle r=2: in (B, 4*co, H, W) -> out (B, co, 2H, 2W)
__global__ void pixel_shuffle2_kernel(const float* __restrict__ in,
                                      float* __restrict__ out, int B, int co,
                                      int H, int W) {
  size_t total = (size_t)B * co * 4 * H * W;
  size_t idx = (size_t)blockIdx.x * blockDim.x + threadIdx.x;
  if (idx >= total) return;
  int W2 = 2 * W;
  size_t perB = (size_t)co * 4 * H * W;
  int b = (int)(idx / perB);
  size_t rem = idx - (size_t)b * perB;
  int ch = (int)(rem / ((size_t)4 * H * W));
  size_t rem2 = rem - (size_t)ch * 4 * H * W;
  int oy = (int)(rem2 / W2);
  int ox = (int)(rem2 - (size_t)oy * W2);
  int y = oy >> 1, ry = oy & 1;
  int x = ox >> 1, rx = ox & 1;
  out[idx] = in[(((size_t)(b * co + ch) * 4) + ry * 2 + rx) * H * W +
                (size_t)y * W + x];
}

// ---------------------------------------------------------------------------
extern "C" void kernel_launch(void* const* d_in, const int* in_sizes, int n_in,
                              void* d_out, int out_size, void* d_ws,
                              size_t ws_size, hipStream_t stream) {
  (void)in_sizes; (void)n_in; (void)out_size; (void)ws_size;
  const int H = 96, W = 96, HW = H * W;

  const int IDX_X = 0, IDX_CF = 1, IDX_RES1 = 3, IDX_CR = 23, IDX_OFF1 = 25,
            IDX_DC1 = 27, IDX_OFF2 = 29, IDX_DC2 = 31, IDX_OFF3 = 33,
            IDX_DC3 = 35, IDX_OFF = 37, IDX_DC = 39, IDX_RECON = 41,
            IDX_RES2 = 43, IDX_FEAEX = 83, IDX_UP1 = 85, IDX_UP2 = 87,
            IDX_UPFIN = 89;
  auto inp = [&](int i) { return (const float*)d_in[i]; };

  // --- workspace layout (fp32 elems), phase-aliased bump allocation ---
  float* WS = (float*)d_ws;
  float* FEATS = WS;                         // 20*64*HW = 11,796,480
  float* TMP   = FEATS + 11796480;           // 11,796,480
  float* CAT   = TMP + 11796480;             // 18,874,368
  float* BIG0  = CAT + 18874368;             // 37,748,736
  float* BIG1  = BIG0 + 37748736;            // 37,748,736
  float* F0 = BIG1;
  float* F1 = BIG1 + 9437184;
  float* SUPP = BIG1 + 18874368;
  float* IM = BIG1 + 28311552;
  float* OFFB = BIG0;
  float* FEA2A = FEATS;
  float* FEA2B = FEATS + 2359296;
  float* UP1O = TMP;
  float* U1 = CAT;
  float* UP2O = BIG0;
  float* U2 = BIG1;

  float* out_hr = (float*)d_out;             // 4*3*384*384 = 1,769,472
  float* LRS = out_hr + 1769472;             // 4*5*3*HW   =   552,960

  auto conv = [&](const float* in, int wi, float* out, int Cin, int Cout,
                  int nImg, int Hh, int Ww, int relu, const float* resid,
                  bool has_bias) {
    dim3 g((Hh * Ww + 63) / 64, (Cout + 63) / 64, nImg);
    conv3x3_wmma<<<g, 256, 0, stream>>>(
        in, inp(wi), has_bias ? inp(wi + 1) : nullptr, resid, out, Cin, Cout,
        Hh, Ww, relu);
  };
  auto dconv = [&](const float* in, const float* off, int wi, float* out) {
    dim3 g((HW + 31) / 32, 1, 16);
    dconv3x3_wmma<<<g, 256, 0, stream>>>(in, off, inp(wi), inp(wi + 1), out, H,
                                         W);
  };
  auto blocks = [&](size_t total) { return (unsigned)((total + 255) / 256); };

  // ================= phase 1: shared feature extraction ====================
  conv(inp(IDX_X), IDX_CF, FEATS, 3, 64, 20, H, W, 1, nullptr, true);
  for (int i = 0; i < 5; ++i) {
    int wi = IDX_RES1 + 4 * i;
    conv(FEATS, wi, TMP, 64, 64, 20, H, W, 1, nullptr, true);
    conv(TMP, wi + 2, FEATS, 64, 64, 20, H, W, 0, FEATS, true);
  }

  // ================= phase 2: PCD-style alignment ==========================
  build_cat_kernel<<<blocks((size_t)16 * 128 * HW), 256, 0, stream>>>(FEATS,
                                                                      CAT, HW);
  build_supp_kernel<<<blocks((size_t)16 * 64 * HW), 256, 0, stream>>>(
      FEATS, SUPP, HW);

  conv(CAT, IDX_CR, F0, 128, 64, 16, H, W, 0, nullptr, true);
  conv(F0, IDX_OFF1, OFFB, 64, 144, 16, H, W, 0, nullptr, true);
  dconv(F0, OFFB, IDX_DC1, F1);
  conv(F1, IDX_OFF2, OFFB, 64, 144, 16, H, W, 0, nullptr, true);
  dconv(F1, OFFB, IDX_DC2, F0);
  conv(F0, IDX_OFF3, OFFB, 64, 144, 16, H, W, 0, nullptr, true);
  dconv(SUPP, OFFB, IDX_DC3, F1);
  conv(F1, IDX_OFF, OFFB, 64, 144, 16, H, W, 0, nullptr, true);
  dconv(F1, OFFB, IDX_DC, F0);
  conv(F0, IDX_RECON, IM, 64, 3, 16, H, W, 0, nullptr, true);

  build_lrs_kernel<<<blocks((size_t)4 * 15 * HW), 256, 0, stream>>>(
      IM, inp(IDX_X), LRS, HW);

  // ================= phase 3: reconstruction + upsampling ==================
  conv(LRS, IDX_FEAEX, FEA2A, 15, 64, 4, H, W, 1, nullptr, true);
  for (int i = 0; i < 10; ++i) {
    int wi = IDX_RES2 + 4 * i;
    conv(FEA2A, wi, FEA2B, 64, 64, 4, H, W, 1, nullptr, true);
    conv(FEA2B, wi + 2, FEA2A, 64, 64, 4, H, W, 0, FEA2A, true);
  }
  conv(FEA2A, IDX_UP1, UP1O, 64, 256, 4, H, W, 0, nullptr, true);
  pixel_shuffle2_kernel<<<blocks((size_t)4 * 256 * HW), 256, 0, stream>>>(
      UP1O, U1, 4, 64, H, W);
  conv(U1, IDX_UP2, UP2O, 64, 256, 4, 192, 192, 0, nullptr, true);
  pixel_shuffle2_kernel<<<blocks((size_t)4 * 256 * 192 * 192), 256, 0,
                          stream>>>(UP2O, U2, 4, 64, 192, 192);
  conv(U2, IDX_UPFIN, out_hr, 64, 3, 4, 384, 384, 0, nullptr, false);
}
